// Encoder_17970143167322
// MI455X (gfx1250) — matrix operand order repaired and checked by reading, not compiled, not run on hardware
//
#include <hip/hip_runtime.h>
#include <hip/hip_bf16.h>

// Shapes: B=512, T-1=127, INPUT=512, HID=256, GATES=1024, K=768 (=512+256)
#define NB    512
#define TS    127
#define NI    512
#define NH    256
#define NG    1024
#define NK    768
#define KSLABS (NK / 32)     // 24 K-slabs of 32

typedef __attribute__((ext_vector_type(16))) _Float16     v16h;
typedef __attribute__((ext_vector_type(8)))  float        v8f;
typedef __attribute__((ext_vector_type(4)))  unsigned int v4u;

union Frag {
    v4u  q[2];     // two 16B chunks -> global_load_b128 pairs
    v16h h;
};

// ---------------------------------------------------------------------------
// Fragment-order indexing (ISA 7.12.2, wave32 f16 WMMA).
// Storage: [tile][kslab][lane][16 halves]; a wave's fragment load is
// 32 lanes x 32B = 1KB contiguous.
// ---------------------------------------------------------------------------
// A 16x32 f16: lanes 0-15 hold K=0..7 & 16..23 of row M=lane;
//              lanes 16-31 hold K=8..15 & 24..31 of row M=lane-16.
__device__ __forceinline__ int a_frag_idx(int row, int k) {
    int mt = row >> 4, r = row & 15;
    int kb = k >> 5,  kk = k & 31;
    int lane = r + (((kk >> 3) & 1) << 4);
    int j = (kk & 7) + ((kk >> 4) << 3);
    return ((mt * KSLABS + kb) * 32 + lane) * 16 + j;
}
// B 32x16 f16: lanes 0-15 hold K=0..7 & 16..23 of col N=lane; lanes 16-31 hold K=8..15 & 24..31.
__device__ __forceinline__ int b_frag_idx(int n, int k) {
    int nt = n >> 4, c = n & 15;
    int kb = k >> 5, kk = k & 31;
    int lane = c + (((kk >> 3) & 1) << 4);
    int j = (kk & 7) + ((kk >> 4) << 3);
    return ((nt * KSLABS + kb) * 32 + lane) * 16 + j;
}

// Branch-free activations (avoid libm tanhf's divergent paths)
__device__ __forceinline__ float sigm(float v) {
    return 1.0f / (1.0f + __expf(-v));
}
__device__ __forceinline__ float tanh_fast(float v) {
    v = fminf(fmaxf(v, -15.0f), 15.0f);
    float e = __expf(2.0f * v);
    return (e - 1.0f) / (e + 1.0f);
}

// ---------------------------------------------------------------------------
// One-time: pack [W_ih | W_hh] (f32) -> Wp2 (f16, fragment order), bias sum
// ---------------------------------------------------------------------------
__global__ void pack_weights_kernel(const float* __restrict__ W_ih,
                                    const float* __restrict__ W_hh,
                                    const float* __restrict__ b_ih,
                                    const float* __restrict__ b_hh,
                                    _Float16* __restrict__ Wp2,
                                    float* __restrict__ bias) {
    int idx = blockIdx.x * 256 + threadIdx.x;           // 0 .. 1024*768-1
    int n = idx / NK;
    int k = idx - n * NK;
    float v = (k < NI) ? W_ih[n * NI + k] : W_hh[n * NH + (k - NI)];
    Wp2[b_frag_idx(n, k)] = (_Float16)v;
    if (idx < NG) bias[idx] = b_ih[idx] + b_hh[idx];
}

// ---------------------------------------------------------------------------
// One-time: score_x[b][i] = sum_t input[b][t][i] * w_x[t] + attn_b
// ---------------------------------------------------------------------------
__global__ void score_x_kernel(const float* __restrict__ x,
                               const float* __restrict__ attn_W,
                               const float* __restrict__ attn_b,
                               float* __restrict__ score_x) {
    int idx = blockIdx.x * 256 + threadIdx.x;           // 0 .. 512*512-1
    int b = idx >> 9;
    int i = idx & 511;
    const float* wx = attn_W + 2 * NH;                  // uniform -> SGPR loads
    float acc = attn_b[0];
    #pragma unroll 4
    for (int t = 0; t < TS; ++t)
        acc += x[(b * TS + t) * NI + i] * wx[t];
    score_x[idx] = acc;
}

// ---------------------------------------------------------------------------
// One-time: zero h, c (f32) and the h-half of the fragment-ordered A operand
// ---------------------------------------------------------------------------
__global__ void init_state_kernel(float* __restrict__ H, float* __restrict__ C,
                                  _Float16* __restrict__ Ap) {
    int idx = blockIdx.x * 256 + threadIdx.x;           // 0 .. 512*256-1
    H[idx] = 0.0f;
    C[idx] = 0.0f;
    int row = idx >> 8, col = idx & 255;
    Ap[a_frag_idx(row, NI + col)] = (_Float16)0.0f;
}

// ---------------------------------------------------------------------------
// Per step: scores -> softmax(512) -> weighted = attn * x_t
// One block (256 threads) per batch row; LDS tree reductions.
// Writes weighted to d_out (f32) and into Ap[:, 0:512] in fragment order.
// ---------------------------------------------------------------------------
__global__ void attn_kernel(const float* __restrict__ x,
                            const float* __restrict__ H,
                            const float* __restrict__ C,
                            const float* __restrict__ attn_W,   // [0:256]=w_h, [256:512]=w_c
                            const float* __restrict__ score_x,
                            float* __restrict__ out_attn,
                            float* __restrict__ out_weighted,
                            _Float16* __restrict__ Ap,
                            int t) {
    __shared__ float red[256];
    int b = blockIdx.x;
    int tid = threadIdx.x;

    // s = h . w_h + c . w_c   (K = 256)
    float p = H[b * NH + tid] * attn_W[tid] + C[b * NH + tid] * attn_W[NH + tid];
    red[tid] = p;
    __syncthreads();
    for (int s = 128; s > 0; s >>= 1) {
        if (tid < s) red[tid] += red[tid + s];
        __syncthreads();
    }
    float sdot = red[0];
    __syncthreads();

    float sc0 = sdot + score_x[b * NI + tid];
    float sc1 = sdot + score_x[b * NI + tid + 256];

    red[tid] = fmaxf(sc0, sc1);
    __syncthreads();
    for (int s = 128; s > 0; s >>= 1) {
        if (tid < s) red[tid] = fmaxf(red[tid], red[tid + s]);
        __syncthreads();
    }
    float mx = red[0];
    __syncthreads();

    float e0 = __expf(sc0 - mx);
    float e1 = __expf(sc1 - mx);
    red[tid] = e0 + e1;
    __syncthreads();
    for (int s = 128; s > 0; s >>= 1) {
        if (tid < s) red[tid] += red[tid + s];
        __syncthreads();
    }
    float inv = 1.0f / red[0];

    float a0 = e0 * inv, a1 = e1 * inv;
    float x0 = x[(b * TS + t) * NI + tid];
    float x1 = x[(b * TS + t) * NI + tid + 256];
    float w0 = a0 * x0, w1 = a1 * x1;

    out_weighted[(b * TS + t) * NI + tid]       = w0;
    out_weighted[(b * TS + t) * NI + tid + 256] = w1;
    Ap[a_frag_idx(b, tid)]       = (_Float16)w0;
    Ap[a_frag_idx(b, tid + 256)] = (_Float16)w1;
    if (t == TS - 1) {
        out_attn[b * NI + tid]       = a0;
        out_attn[b * NI + tid + 256] = a1;
    }
}

// ---------------------------------------------------------------------------
// Per step: fused gates-GEMM + LSTM cell.
// One wave per (16-row batch tile, 16-col gate tile): 4 accumulators
// (i/f/g/o at N, N+256, N+512, N+768) share one A fragment per K-slab
// (4x A reuse); all fragment loads are 1KB-contiguous b128s. The cell
// epilogue is per-lane register math (identical C/D slot layout across
// the 4 accumulators) -- no LDS, no barrier.
// Grid: 64 blocks x 256 threads = 512 waves = 32 Mtiles x 16 Ntiles.
// ---------------------------------------------------------------------------
__global__ void gemm_cell_kernel(_Float16* __restrict__ Ap,       // frag-ordered 512x768
                                 const _Float16* __restrict__ Wp2,// frag-ordered 1024x768
                                 const float* __restrict__ bias,  // 1024
                                 float* __restrict__ H,
                                 float* __restrict__ C,
                                 float* __restrict__ out_enc,     // [b][t][j]
                                 int t) {
    int wave = threadIdx.x >> 5;
    int lane = threadIdx.x & 31;
    int wid  = blockIdx.x * 8 + wave;       // 0..511
    int m    = wid >> 4;                    // batch tile 0..31
    int ng   = wid & 15;                    // gate-column tile 0..15
    int hi   = lane >> 4;
    int ln   = lane & 15;

    // fragment stream bases: each K-slab advances 32 lanes * 16 halves = 512
    const _Float16* ap  = Ap  + ((m * KSLABS) * 32 + lane) * 16;
    const _Float16* bp0 = Wp2 + (((0 * 16 + ng) * KSLABS) * 32 + lane) * 16;
    const _Float16* bp1 = Wp2 + (((1 * 16 + ng) * KSLABS) * 32 + lane) * 16;
    const _Float16* bp2 = Wp2 + (((2 * 16 + ng) * KSLABS) * 32 + lane) * 16;
    const _Float16* bp3 = Wp2 + (((3 * 16 + ng) * KSLABS) * 32 + lane) * 16;

    v8f acc0 = {}, acc1 = {}, acc2 = {}, acc3 = {};

    for (int s = 0; s < KSLABS; ++s) {
        int off = s * 512;                               // halves per K-slab
        __builtin_prefetch(ap + off + 512, 0, 3);        // next A slab, keep in cache
        Frag fa, fb0, fb1, fb2, fb3;
        fa.q[0]  = *(const v4u*)(ap  + off);  fa.q[1]  = *(const v4u*)(ap  + off + 8);
        fb0.q[0] = *(const v4u*)(bp0 + off);  fb0.q[1] = *(const v4u*)(bp0 + off + 8);
        fb1.q[0] = *(const v4u*)(bp1 + off);  fb1.q[1] = *(const v4u*)(bp1 + off + 8);
        fb2.q[0] = *(const v4u*)(bp2 + off);  fb2.q[1] = *(const v4u*)(bp2 + off + 8);
        fb3.q[0] = *(const v4u*)(bp3 + off);  fb3.q[1] = *(const v4u*)(bp3 + off + 8);
        acc0 = __builtin_amdgcn_wmma_f32_16x16x32_f16(false, fa.h, false, fb0.h,
                                                      (short)0, acc0, false, false);
        acc1 = __builtin_amdgcn_wmma_f32_16x16x32_f16(false, fa.h, false, fb1.h,
                                                      (short)0, acc1, false, false);
        acc2 = __builtin_amdgcn_wmma_f32_16x16x32_f16(false, fa.h, false, fb2.h,
                                                      (short)0, acc2, false, false);
        acc3 = __builtin_amdgcn_wmma_f32_16x16x32_f16(false, fa.h, false, fb3.h,
                                                      (short)0, acc3, false, false);
    }

    int j = ng * 16 + ln;                                // output column (per lane)
    float bi = bias[j], bf = bias[j + NH], bg = bias[j + 2*NH], bo = bias[j + 3*NH];

    // C/D f32 layout: VGPR r -> row r + 8*hi ; col = lane&15
    #pragma unroll
    for (int r = 0; r < 8; ++r) {
        int row = m * 16 + r + (hi << 3);                // batch index
        float iv = acc0[r] + bi;
        float fv = acc1[r] + bf;
        float gv = acc2[r] + bg;
        float ov = acc3[r] + bo;
        float c_old = C[row * NH + j];
        float c_new = sigm(fv) * c_old + sigm(iv) * tanh_fast(gv);
        float h_new = sigm(ov) * tanh_fast(c_new);
        C[row * NH + j] = c_new;
        H[row * NH + j] = h_new;
        Ap[a_frag_idx(row, NI + j)] = (_Float16)h_new;   // h half of next step's A
        out_enc[(row * TS + t) * NH + j] = h_new;
    }
}

// ---------------------------------------------------------------------------
extern "C" void kernel_launch(void* const* d_in, const int* in_sizes, int n_in,
                              void* d_out, int out_size, void* d_ws, size_t ws_size,
                              hipStream_t stream) {
    const float* x      = (const float*)d_in[0];   // (512,127,512)
    const float* W_ih   = (const float*)d_in[1];   // (1024,512)
    const float* W_hh   = (const float*)d_in[2];   // (1024,256)
    const float* b_ih   = (const float*)d_in[3];   // (1024,)
    const float* b_hh   = (const float*)d_in[4];   // (1024,)
    const float* attn_W = (const float*)d_in[5];   // (1,639)
    const float* attn_b = (const float*)d_in[6];   // (1,)

    // workspace layout (byte offsets, all 32B aligned)
    char* ws = (char*)d_ws;
    _Float16* Wp2     = (_Float16*)(ws);                      // 1024*768*2 = 1572864
    float*    bias    = (float*)   (ws + 1572864);            // 1024*4     = 4096
    float*    score_x = (float*)   (ws + 1576960);            // 512*512*4  = 1048576
    _Float16* Ap      = (_Float16*)(ws + 2625536);            // 512*768*2  = 786432
    float*    H       = (float*)   (ws + 3411968);            // 512*256*4  = 524288
    float*    C       = (float*)   (ws + 3936256);            // 512*256*4  = 524288

    // d_out: attn_weights (512*512) | input_weighted (512*127*512) | input_encoded (512*127*256)
    float* out_attn = (float*)d_out;
    float* out_w    = out_attn + NB * NI;
    float* out_enc  = out_w + (size_t)NB * TS * NI;

    pack_weights_kernel<<<(NG * NK) / 256, 256, 0, stream>>>(W_ih, W_hh, b_ih, b_hh, Wp2, bias);
    score_x_kernel<<<(NB * NI) / 256, 256, 0, stream>>>(x, attn_W, attn_b, score_x);
    init_state_kernel<<<(NB * NH) / 256, 256, 0, stream>>>(H, C, Ap);

    for (int t = 0; t < TS; ++t) {
        attn_kernel<<<NB, 256, 0, stream>>>(x, H, C, attn_W, score_x,
                                            out_attn, out_w, Ap, t);
        gemm_cell_kernel<<<64, 256, 0, stream>>>(Ap, Wp2, bias, H, C, out_enc, t);
    }
}
